// GNN_block_56023553409507
// MI455X (gfx1250) — matrix operand order repaired and checked by reading, not compiled, not run on hardware
//
#include <hip/hip_runtime.h>

typedef _Float16 h16;
typedef __attribute__((ext_vector_type(16))) _Float16 v16h;
typedef __attribute__((ext_vector_type(8)))  _Float16 v8h;
typedef __attribute__((ext_vector_type(8)))  float    v8f;

#define NN      16384       // B*L nodes
#define DM      256         // d_model
#define NHEAD   4
#define CH      256         // per-head channels
#define HC      1024        // NHEAD*CH
#define NE      65536       // edges
#define ETOT    81920       // edges + self loops
#define NEG_SLOPE 0.2f
#define LN_EPS  1e-6f

// ---------------------------------------------------------------- LayerNorm -> f16
__global__ void ln_kernel(const float* __restrict__ x, const float* __restrict__ gamma,
                          const float* __restrict__ beta, h16* __restrict__ xln) {
  int n = blockIdx.x;
  int t = threadIdx.x;
  __shared__ float red[256];
  float v = x[n * DM + t];
  red[t] = v; __syncthreads();
  for (int s = 128; s > 0; s >>= 1) { if (t < s) red[t] += red[t + s]; __syncthreads(); }
  float mu = red[0] * (1.0f / DM); __syncthreads();
  float dv = v - mu;
  red[t] = dv * dv; __syncthreads();
  for (int s = 128; s > 0; s >>= 1) { if (t < s) red[t] += red[t + s]; __syncthreads(); }
  float var = red[0] * (1.0f / DM);
  float y = dv * rsqrtf(var + LN_EPS) * gamma[t] + beta[t];
  xln[n * DM + t] = (h16)y;
}

// ---------------------------------------------------------------- f32 -> f16 convert
__global__ void cvt_kernel(const float* __restrict__ src, h16* __restrict__ dst, int n) {
  for (int i = blockIdx.x * blockDim.x + threadIdx.x; i < n; i += gridDim.x * blockDim.x)
    dst[i] = (h16)src[i];
}

// ---------------------------------------------------------------- fill f32
__global__ void fill_kernel(float* __restrict__ p, float val, int n) {
  for (int i = blockIdx.x * blockDim.x + threadIdx.x; i < n; i += gridDim.x * blockDim.x)
    p[i] = val;
}

// ---------------------------------------------------------------- WMMA GEMM
// C[M,Nc] = A[M,K] @ W[Nc,K]^T (+bias) (+residual).  One wave = one 16x16 tile.
// mode 0: outF = acc (+bias)      mode 1: outH = (h16)(acc + bias)
// mode 2: outF = acc + bias + resid
__global__ void gemm_wmma(const h16* __restrict__ A, const h16* __restrict__ W,
                          const float* __restrict__ bias, int M, int Nc, int K,
                          float* __restrict__ outF, h16* __restrict__ outH,
                          const float* __restrict__ resid, int mode) {
  int wave = (blockIdx.x * blockDim.x + threadIdx.x) >> 5;
  int lane = threadIdx.x & 31;
  int tiles_n = Nc >> 4;
  int tiles = (M >> 4) * tiles_n;
  if (wave >= tiles) return;
  int tm = wave / tiles_n, tn = wave % tiles_n;
  int row0 = tm << 4, col0 = tn << 4;

  int hi = lane >> 4;          // half-wave select
  int arow = row0 + (lane & 15);
  int bcol = col0 + (lane & 15);
  const h16* Ap = A + (long)arow * K;
  const h16* Wp = W + (long)bcol * K;

  v8f acc = {};
  for (int kb = 0; kb < K; kb += 32) {
    // A fragment 16x32 f16: lane<16 holds K {kb+0..7, kb+16..23}; lane>=16 +8
    v8h a_lo = *(const v8h*)(Ap + kb + hi * 8);
    v8h a_hi = *(const v8h*)(Ap + kb + 16 + hi * 8);
    // B fragment 32x16 f16: lane<16 holds K {kb+0..7, kb+16..23} of column bcol; lane>=16 +8
    v8h b_lo = *(const v8h*)(Wp + kb + hi * 8);
    v8h b_hi = *(const v8h*)(Wp + kb + 16 + hi * 8);
    if (kb + 32 < K) {
      __builtin_prefetch(Ap + kb + 32, 0, 1);
      __builtin_prefetch(Wp + kb + 32, 0, 1);
    }
    v16h a, b;
    for (int i = 0; i < 8; ++i) { a[i] = a_lo[i]; a[8 + i] = a_hi[i]; }
    for (int i = 0; i < 8; ++i) { b[i] = b_lo[i]; b[8 + i] = b_hi[i]; }
    acc = __builtin_amdgcn_wmma_f32_16x16x32_f16(
        /*neg_a=*/false, a, /*neg_b=*/false, b,
        /*c_mod=*/(short)0, acc, /*reuse_a=*/false, /*reuse_b=*/false);
  }

  // D layout: lanes 0-15 -> M = vgpr, N = lane; lanes 16-31 -> M = 8+vgpr, N = lane-16
  int ocol = col0 + (lane & 15);
  float bv = bias ? bias[ocol] : 0.0f;
  for (int v = 0; v < 8; ++v) {
    int orow = row0 + v + 8 * hi;
    long idx = (long)orow * Nc + ocol;
    float r = acc[v] + bv;
    if (mode == 0)      outF[idx] = r;
    else if (mode == 1) outH[idx] = (h16)r;
    else                outF[idx] = r + resid[idx];
  }
}

// ---------------------------------------------------------------- attention dots
// a_src[n,h] = sum_c xh[n,h,c]*att_src[h,c]; same for dst. One wave per (n,h).
__global__ void att_kernel(const float* __restrict__ xh, const float* __restrict__ att_s,
                           const float* __restrict__ att_d,
                           float* __restrict__ a_src, float* __restrict__ a_dst) {
  int w = (blockIdx.x * blockDim.x + threadIdx.x) >> 5;
  int lane = threadIdx.x & 31;
  if (w >= NN * NHEAD) return;
  int n = w >> 2, h = w & 3;
  const float* xp = xh + (long)n * HC + h * CH;
  const float* as = att_s + h * CH;
  const float* ad = att_d + h * CH;
  float s1 = 0.f, s2 = 0.f;
  for (int c = lane; c < CH; c += 32) { float xv = xp[c]; s1 += xv * as[c]; s2 += xv * ad[c]; }
  for (int off = 16; off; off >>= 1) { s1 += __shfl_xor(s1, off, 32); s2 += __shfl_xor(s2, off, 32); }
  if (lane == 0) { a_src[w] = s1; a_dst[w] = s2; }
}

__device__ __forceinline__ void atomicMaxF(float* addr, float val) {
  if (val >= 0.f) atomicMax((int*)addr, __float_as_int(val));
  else            atomicMin((unsigned int*)addr, __float_as_uint(val));
}

__device__ __forceinline__ void edge_nodes(const int* __restrict__ ei, int t, int& s, int& d) {
  if (t < NE) { s = ei[t]; d = ei[NE + t]; }
  else        { s = d = t - NE; }        // self loop
}

// ---------------------------------------------------------------- segment max of leaky-relu logits
__global__ void edge_max_kernel(const int* __restrict__ ei, const float* __restrict__ a_src,
                                const float* __restrict__ a_dst, float* __restrict__ emax) {
  int t = blockIdx.x * blockDim.x + threadIdx.x;
  if (t >= ETOT) return;
  int s, d; edge_nodes(ei, t, s, d);
  for (int h = 0; h < NHEAD; ++h) {
    float l = a_src[s * NHEAD + h] + a_dst[d * NHEAD + h];
    l = (l > 0.f) ? l : NEG_SLOPE * l;
    atomicMaxF(&emax[d * NHEAD + h], l);
  }
}

// ---------------------------------------------------------------- exp + segment sum
__global__ void edge_exp_kernel(const int* __restrict__ ei, const float* __restrict__ a_src,
                                const float* __restrict__ a_dst, const float* __restrict__ emax,
                                float* __restrict__ ebuf, float* __restrict__ denom) {
  int t = blockIdx.x * blockDim.x + threadIdx.x;
  if (t >= ETOT) return;
  int s, d; edge_nodes(ei, t, s, d);
  for (int h = 0; h < NHEAD; ++h) {
    float l = a_src[s * NHEAD + h] + a_dst[d * NHEAD + h];
    l = (l > 0.f) ? l : NEG_SLOPE * l;
    float e = __expf(l - emax[d * NHEAD + h]);
    ebuf[t * NHEAD + h] = e;
    atomicAdd(&denom[d * NHEAD + h], e);
  }
}

// ---------------------------------------------------------------- weighted message scatter (1 wave / edge)
__global__ void edge_scatter_kernel(const int* __restrict__ ei, const float* __restrict__ xh,
                                    const float* __restrict__ ebuf, const float* __restrict__ denom,
                                    float* __restrict__ agg) {
  int w = (blockIdx.x * blockDim.x + threadIdx.x) >> 5;
  int lane = threadIdx.x & 31;
  if (w >= ETOT) return;
  int s, d; edge_nodes(ei, w, s, d);
  for (int h = 0; h < NHEAD; ++h) {
    float alpha = ebuf[w * NHEAD + h] / denom[d * NHEAD + h];
    const float* xp = xh + (long)s * HC + h * CH;
    float*       op = agg + (long)d * HC + h * CH;
    for (int c = lane; c < CH; c += 32) atomicAdd(&op[c], xp[c] * alpha);
  }
}

// ---------------------------------------------------------------- head mean + bias -> f16
__global__ void mean_kernel(const float* __restrict__ agg, const float* __restrict__ bias,
                            h16* __restrict__ gat) {
  int i = blockIdx.x * blockDim.x + threadIdx.x;
  if (i >= NN * CH) return;
  int n = i >> 8, c = i & 255;
  const float* p = agg + (long)n * HC;
  float v = 0.25f * (p[c] + p[CH + c] + p[2 * CH + c] + p[3 * CH + c]) + bias[c];
  gat[i] = (h16)v;
}

// ================================================================ launch
extern "C" void kernel_launch(void* const* d_in, const int* in_sizes, int n_in,
                              void* d_out, int out_size, void* d_ws, size_t ws_size,
                              hipStream_t stream) {
  const float* inputs   = (const float*)d_in[0];
  const int*   eindex   = (const int*)  d_in[1];
  const float* ln_gamma = (const float*)d_in[2];
  const float* ln_beta  = (const float*)d_in[3];
  const float* W1       = (const float*)d_in[4];
  const float* b1       = (const float*)d_in[5];
  const float* W_gat    = (const float*)d_in[6];
  const float* att_src  = (const float*)d_in[7];
  const float* att_dst  = (const float*)d_in[8];
  const float* bias_gat = (const float*)d_in[9];
  const float* W2       = (const float*)d_in[10];
  const float* b2       = (const float*)d_in[11];
  float* out = (float*)d_out;

  // workspace bump allocator (256B aligned)
  char* base = (char*)d_ws;
  size_t off = 0;
  auto alloc = [&](size_t bytes) { void* p = base + off; off = (off + bytes + 255) & ~(size_t)255; return p; };
  h16*   xln    = (h16*)  alloc((size_t)NN * DM * 2);
  h16*   W1h    = (h16*)  alloc((size_t)DM * DM * 2);
  h16*   Wgath  = (h16*)  alloc((size_t)HC * DM * 2);
  h16*   W2h    = (h16*)  alloc((size_t)DM * DM * 2);
  h16*   gnn_in = (h16*)  alloc((size_t)NN * DM * 2);
  float* xh     = (float*)alloc((size_t)NN * HC * 4);
  float* a_src  = (float*)alloc((size_t)NN * NHEAD * 4);
  float* a_dst  = (float*)alloc((size_t)NN * NHEAD * 4);
  float* emax   = (float*)alloc((size_t)NN * NHEAD * 4);
  float* denom  = (float*)alloc((size_t)NN * NHEAD * 4);
  float* ebuf   = (float*)alloc((size_t)ETOT * NHEAD * 4);
  float* agg    = (float*)alloc((size_t)NN * HC * 4);
  h16*   gat    = (h16*)  alloc((size_t)NN * DM * 2);
  (void)ws_size; (void)in_sizes; (void)n_in; (void)out_size;

  // 1. LayerNorm -> f16
  ln_kernel<<<NN, 256, 0, stream>>>(inputs, ln_gamma, ln_beta, xln);

  // 2. weight converts
  cvt_kernel<<<256, 256, 0, stream>>>(W1,    W1h,   DM * DM);
  cvt_kernel<<<1024, 256, 0, stream>>>(W_gat, Wgath, HC * DM);
  cvt_kernel<<<256, 256, 0, stream>>>(W2,    W2h,   DM * DM);

  // 3. gnn_in = xln @ W1^T + b1   -> f16   (tiles: (16384/16)*(256/16) = 16384, 8 waves/block)
  gemm_wmma<<<16384 / 8, 256, 0, stream>>>(xln, W1h, b1, NN, DM, DM,
                                           nullptr, gnn_in, nullptr, 1);

  // 4. xh = gnn_in @ W_gat^T      -> f32   (tiles: 1024*64 = 65536)
  gemm_wmma<<<65536 / 8, 256, 0, stream>>>(gnn_in, Wgath, nullptr, NN, HC, DM,
                                           xh, nullptr, nullptr, 0);

  // 5. per-node attention logits (one wave per (n,h): 65536 waves)
  att_kernel<<<(NN * NHEAD) / 8, 256, 0, stream>>>(xh, att_src, att_dst, a_src, a_dst);

  // 6. init segment buffers
  fill_kernel<<<256, 256, 0, stream>>>(emax,  -__builtin_huge_valf(), NN * NHEAD);
  fill_kernel<<<256, 256, 0, stream>>>(denom, 0.0f, NN * NHEAD);
  fill_kernel<<<8192, 256, 0, stream>>>(agg,  0.0f, NN * HC);

  // 7. segment softmax over destination nodes
  edge_max_kernel<<<ETOT / 256, 256, 0, stream>>>(eindex, a_src, a_dst, emax);
  edge_exp_kernel<<<ETOT / 256, 256, 0, stream>>>(eindex, a_src, a_dst, emax, ebuf, denom);

  // 8. weighted message aggregation (one wave per edge: 81920 waves)
  edge_scatter_kernel<<<ETOT / 8, 256, 0, stream>>>(eindex, xh, ebuf, denom, agg);

  // 9. head mean + bias -> f16
  mean_kernel<<<(NN * CH) / 256, 256, 0, stream>>>(agg, bias_gat, gat);

  // 10. y = gat @ W2^T + b2 + inputs  -> d_out
  gemm_wmma<<<16384 / 8, 256, 0, stream>>>(gat, W2h, b2, NN, DM, DM,
                                           out, nullptr, inputs, 2);
}
